// mgfn_52201032515884
// MI455X (gfx1250) — matrix operand order repaired and checked by reading, not compiled, not run on hardware
//
#include <hip/hip_runtime.h>
#include <hip/hip_bf16.h>
#include <utility>

typedef __attribute__((ext_vector_type(16))) __bf16 v16bf;
typedef __attribute__((ext_vector_type(8)))  __bf16 v8bf;
typedef __attribute__((ext_vector_type(2)))  __bf16 v2bf;
typedef __attribute__((ext_vector_type(8)))  float  v8f;

#define EPSF 1e-5f

// ------------------------------------------------------------------
// conv1d-as-GEMM, bf16 WMMA, fp32 accumulate, double-buffered LDS.
//   C[b, m, 0..63] = sum_tap A_tap[m,:] @ B[b, :, n+tap-pad]
//   A[m,k,tap] : A + m*aRS + k*aCS + tap*aTS     (fp32 weights)
//   B[b,k,n]   : Bm + b*bBS + k*bKS + n*bNS      (fp32 activations)
// Host guarantees: M % 64 == 0, K % 32 == 0, N == 64.
// 128 threads = 4 waves; block tile 64M x 64N; each wave owns 16M x 64N
// (4 accumulators); A fragment loaded once per k-chunk, reused 4x.
// Staging packs 2 bf16 per 32-bit LDS store; chunk c+1 staged while
// chunk c computes (software pipeline, 1 barrier/iter).
// ------------------------------------------------------------------
__device__ __forceinline__ void stage_tiles(
    __bf16* __restrict__ aT, __bf16* __restrict__ bT,
    const float* __restrict__ At, int aRS, int aCS,
    const float* __restrict__ Bb, int bKS, int bNS,
    int k0, int shift, int tid)
{
    // A: 64m x 32k chunk as 1024 packed pairs
    if (aCS == 1) {                       // unit-stride weights: 8-byte loads
        const float* Ak = At + k0;
        #pragma unroll
        for (int i = 0; i < 8; ++i) {
            int pp = tid + i * 128;
            int m = pp >> 4, kp = (pp & 15) << 1;
            float2 f2 = *(const float2*)&Ak[m * aRS + kp];
            v2bf p; p[0] = (__bf16)f2.x; p[1] = (__bf16)f2.y;
            *(v2bf*)&aT[m * 32 + kp] = p;
        }
    } else {
        #pragma unroll
        for (int i = 0; i < 8; ++i) {
            int pp = tid + i * 128;
            int m = pp >> 4, kp = (pp & 15) << 1;
            float v0 = At[m * aRS + (k0 + kp) * aCS];
            float v1 = At[m * aRS + (k0 + kp + 1) * aCS];
            v2bf p; p[0] = (__bf16)v0; p[1] = (__bf16)v1;
            *(v2bf*)&aT[m * 32 + kp] = p;
        }
    }
    // B: 32k x 64n chunk as 1024 packed pairs, branch-free pad handling
    #pragma unroll
    for (int i = 0; i < 8; ++i) {
        int pp = tid + i * 128;
        int n = pp & 63, kp = (pp >> 6) << 1;
        int gn  = n + shift;
        int gnc = gn < 0 ? 0 : (gn > 63 ? 63 : gn);
        const float* bp = Bb + gnc * bNS + (k0 + kp) * bKS;
        float v0 = bp[0];
        float v1 = bp[bKS];
        bool ok = (gn == gnc);
        v2bf p; p[0] = (__bf16)(ok ? v0 : 0.f); p[1] = (__bf16)(ok ? v1 : 0.f);
        *(v2bf*)&bT[n * 32 + kp] = p;
    }
}

__global__ __launch_bounds__(128)
void k_gemm_wmma(const float* __restrict__ A, int aRS, int aCS, int aTS,
                 int taps, int pad,
                 const float* __restrict__ Bm, int bKS, int bNS, int bBS,
                 const float* __restrict__ bias,
                 const float* __restrict__ addSrc, int addBS,
                 float* __restrict__ C, int cBS,
                 int M, int K, float outScale, int actGelu)
{
    __shared__ __align__(64) __bf16 aTile[2][64 * 32];   // [m][kk]
    __shared__ __align__(64) __bf16 bTile[2][64 * 32];   // [n][kk]
    const int tid   = threadIdx.x;
    const int wave  = tid >> 5;
    const int lane  = tid & 31;
    const int g     = lane >> 4;
    const int lq    = lane & 15;
    const int m0    = blockIdx.x * 64;
    const int batch = blockIdx.y;

    const float* Bb = Bm + (long long)batch * bBS;
    const float* A0 = A + (long long)m0 * aRS;

    v8f acc0 = {0.f,0.f,0.f,0.f,0.f,0.f,0.f,0.f};
    v8f acc1 = acc0, acc2 = acc0, acc3 = acc0;

    const int NC = taps * (K >> 5);
    // prologue: stage chunk 0 (tap 0, k0 0)
    stage_tiles(aTile[0], bTile[0], A0, aRS, aCS, Bb, bKS, bNS, 0, -pad, tid);
    int tapN = 0, k0N = 0;   // coords of the most recently staged chunk

    const int arow = (wave * 16 + lq) * 32;
    const int brow = lq * 32 + 16 * g;

    for (int c = 0; c < NC; ++c) {
        const int p = c & 1;
        __syncthreads();
        if (c + 1 < NC) {
            k0N += 32;
            if (k0N == K) { k0N = 0; ++tapN; }
            stage_tiles(aTile[1 - p], bTile[1 - p],
                        A0 + tapN * aTS, aRS, aCS, Bb, bKS, bNS,
                        k0N, tapN - pad, tid);
        }
        // ---- A fragment (contiguous k order, k = 16*g + s): two adjacent 16B runs
        const __bf16* aT = aTile[p];
        const __bf16* bT = bTile[p];
        union { v16bf v; v8bf h[2]; } au;
        au.h[0] = *(const v8bf*)(&aT[arow + 16 * g]);
        au.h[1] = *(const v8bf*)(&aT[arow + 16 * g + 8]);
        // ---- 4 B fragments (contiguous 32B each), 4 WMMAs reuse A
        v16bf b0 = *(const v16bf*)(&bT[brow]);
        v16bf b1 = *(const v16bf*)(&bT[brow + 16 * 32]);
        v16bf b2 = *(const v16bf*)(&bT[brow + 32 * 32]);
        v16bf b3 = *(const v16bf*)(&bT[brow + 48 * 32]);
        acc0 = __builtin_amdgcn_wmma_f32_16x16x32_bf16(false, au.v, false, b0, (short)0, acc0, false, false);
        acc1 = __builtin_amdgcn_wmma_f32_16x16x32_bf16(false, au.v, false, b1, (short)0, acc1, false, false);
        acc2 = __builtin_amdgcn_wmma_f32_16x16x32_bf16(false, au.v, false, b2, (short)0, acc2, false, false);
        acc3 = __builtin_amdgcn_wmma_f32_16x16x32_bf16(false, au.v, false, b3, (short)0, acc3, false, false);
    }

    // ---- epilogue: D layout lanes 0-15 -> M=r, lanes 16-31 -> M=8+r
    const int mw = m0 + wave * 16;
    const long long cb = (long long)batch * cBS;
    const long long ab = (long long)batch * addBS;
    #pragma unroll
    for (int nt = 0; nt < 4; ++nt) {
        v8f a = (nt == 0) ? acc0 : (nt == 1) ? acc1 : (nt == 2) ? acc2 : acc3;
        int n = nt * 16 + lq;
        #pragma unroll
        for (int r = 0; r < 8; ++r) {
            int m = mw + g * 8 + r;
            float v = a[r];
            if (bias) v += bias[m];
            if (actGelu) v = 0.5f * v * (1.f + erff(v * 0.70710678118f));
            v *= outScale;
            if (addSrc) v += addSrc[ab + m * 64 + n];
            C[cb + m * 64 + n] = v;
        }
    }
}

// x[b,co,t] += 0.1 * (to_mag conv: 1 input channel, k=3, pad=1)
__global__ __launch_bounds__(256)
void k_magadd(const float* __restrict__ video, const float* __restrict__ w,
              const float* __restrict__ bias, float* __restrict__ x, int total)
{
    int i = blockIdx.x * 256 + threadIdx.x;
    if (i >= total) return;
    int t  = i & 63;
    int co = (i >> 6) & 63;
    int bb = i >> 12;
    const float* vm = video + (long long)bb * (64 * 2049) + 2048;
    float s = bias[co];
    #pragma unroll
    for (int k = 0; k < 3; ++k) {
        int tt = t + k - 1;
        if (tt >= 0 && tt < 64) s += w[co * 3 + k] * vm[tt * 2049];
    }
    x[i] += 0.1f * s;
}

// ------------------------------------------------------------------
// LayerNorm over channels, per (b,t).
// mode 0: (x-mu)/(sqrt(var)+eps)*g+b -> [b,c,t] ; mode 1: (x-mu)/sqrt(var+eps)*g+b -> [b,t,c]
// ------------------------------------------------------------------
__global__ __launch_bounds__(256)
void k_layernorm(const float* __restrict__ x, const float* __restrict__ g,
                 const float* __restrict__ b, float* __restrict__ out,
                 int C, int T, int mode)
{
    __shared__ float red[256];
    int bt = blockIdx.x;
    int bb = bt / T, t = bt - bb * T;
    const float* xp = x + (long long)bb * C * T + t;
    int tid = threadIdx.x;

    float s = 0.f;
    for (int c = tid; c < C; c += 256) s += xp[c * T];
    red[tid] = s; __syncthreads();
    for (int o = 128; o > 0; o >>= 1) { if (tid < o) red[tid] += red[tid + o]; __syncthreads(); }
    float mean = red[0] / C;
    __syncthreads();

    float s2 = 0.f;
    for (int c = tid; c < C; c += 256) { float d = xp[c * T] - mean; s2 += d * d; }
    red[tid] = s2; __syncthreads();
    for (int o = 128; o > 0; o >>= 1) { if (tid < o) red[tid] += red[tid + o]; __syncthreads(); }
    float var = red[0] / C;
    float inv = (mode == 0) ? (1.f / (sqrtf(var) + EPSF)) : rsqrtf(var + EPSF);

    for (int c = tid; c < C; c += 256) {
        float v = (xp[c * T] - mean) * inv * g[c] + b[c];
        if (mode == 0) out[(long long)bb * C * T + c * T + t] = v;
        else           out[(long long)bb * T * C + t * C + c] = v;
    }
}

// BatchNorm (inference) elementwise
__global__ __launch_bounds__(256)
void k_bn(const float* __restrict__ x, const float* __restrict__ g, const float* __restrict__ b,
          const float* __restrict__ rm, const float* __restrict__ rv,
          float* __restrict__ out, int C, int T, long long total)
{
    long long i = (long long)blockIdx.x * 256 + threadIdx.x;
    if (i >= total) return;
    int c = (int)((i / T) % C);
    out[i] = (x[i] - rm[c]) * rsqrtf(rv[c] + EPSF) * g[c] + b[c];
}

// Depthwise "focus" rel-conv: channel c uses head h = c % heads, KERNEL=5, pad=2
__global__ __launch_bounds__(256)
void k_dw(const float* __restrict__ x, const float* __restrict__ w, const float* __restrict__ bias,
          float* __restrict__ out, int C, int T, int heads, long long total)
{
    long long i = (long long)blockIdx.x * 256 + threadIdx.x;
    if (i >= total) return;
    int t = (int)(i % T);
    int c = (int)((i / T) % C);
    int h = c % heads;
    float s = bias[h];
    #pragma unroll
    for (int k = 0; k < 5; ++k) {
        int tt = t + k - 2;
        if (tt >= 0 && tt < T) s += w[h * 5 + k] * x[i + (k - 2)];
    }
    out[i] = s;
}

// Stage-0 glance attention: heads=1, dim_head=64, T=64. One block per batch.
__global__ __launch_bounds__(64)
void k_glance_attn(const float* __restrict__ qkv, float* __restrict__ out)
{
    __shared__ float qs[4096];  // [d][t], q pre-scaled
    __shared__ float ks[4096];
    __shared__ float vs[4096];
    int b = blockIdx.x, i = threadIdx.x;
    const float* base = qkv + (long long)b * 192 * 64;
    for (int e = i; e < 4096; e += 64) {
        qs[e] = base[e] * 0.125f;     // dim_head^-0.5
        ks[e] = base[4096 + e];
        vs[e] = base[8192 + e];
    }
    __syncthreads();

    float sv[64];
    float mx = -1e30f;
    for (int j = 0; j < 64; ++j) {
        float s = 0.f;
        for (int d = 0; d < 64; ++d) s += qs[d * 64 + i] * ks[d * 64 + j];
        sv[j] = s; mx = fmaxf(mx, s);
    }
    float sum = 0.f;
    for (int j = 0; j < 64; ++j) { sv[j] = expf(sv[j] - mx); sum += sv[j]; }
    float r = 1.f / sum;
    __syncthreads();
    for (int j = 0; j < 64; ++j) qs[i * 64 + j] = sv[j] * r;   // reuse qs as attn[i][j]
    __syncthreads();

    float* op = out + (long long)b * 4096;
    for (int d = 0; d < 64; ++d) {
        float s = 0.f;
        for (int j = 0; j < 64; ++j) s += qs[i * 64 + j] * vs[d * 64 + j];
        op[d * 64 + i] = s;           // out_c[b, d, i]
    }
}

// Fused per-(bc,t): sigmoid(xl . fc_w + fc_b) and ||xl||
__global__ __launch_bounds__(256)
void k_score_mag(const float* __restrict__ xl, const float* __restrict__ fcw,
                 const float* __restrict__ fcb,
                 float* __restrict__ score_raw, float* __restrict__ mag_raw, int F)
{
    __shared__ float red[256];
    long long row = blockIdx.x;
    const float* xp = xl + row * F;
    int tid = threadIdx.x;
    float dot = 0.f, sq = 0.f;
    for (int f = tid; f < F; f += 256) { float v = xp[f]; dot += v * fcw[f]; sq += v * v; }
    red[tid] = dot; __syncthreads();
    for (int o = 128; o > 0; o >>= 1) { if (tid < o) red[tid] += red[tid + o]; __syncthreads(); }
    float dtot = red[0]; __syncthreads();
    red[tid] = sq; __syncthreads();
    for (int o = 128; o > 0; o >>= 1) { if (tid < o) red[tid] += red[tid + o]; __syncthreads(); }
    if (tid == 0) {
        score_raw[row] = 1.f / (1.f + expf(-(dtot + fcb[0])));
        mag_raw[row]   = sqrtf(red[0]);
    }
}

// out layout: score_abn[8] | score_nor[8] | abn_fm[245760] | nor_fm[245760] | scores[1024]
#define OUT_SCORES 491536
__global__ __launch_bounds__(64)
void k_msnsd(const float* __restrict__ score_raw, const float* __restrict__ mag_raw,
             float* __restrict__ out, int* __restrict__ idxbuf)
{
    __shared__ float s[1024], mg[1024];
    __shared__ int idx_sh[48];
    int tid = threadIdx.x;
    for (int e = tid; e < 1024; e += 64) {
        int b = e >> 6, t = e & 63;
        float as = 0.f, am = 0.f;
        for (int c = 0; c < 10; ++c) {
            int r = (b * 10 + c) * 64 + t;
            as += score_raw[r]; am += mag_raw[r];
        }
        s[e] = as * 0.1f; mg[e] = am * 0.1f;
        out[OUT_SCORES + e] = s[e];                 // scores (16,64,1)
    }
    __syncthreads();
    if (tid < 16) {                                  // top-3 per magnitude row
        const float* row = mg + tid * 64;
        int chosen[3];
        for (int k = 0; k < 3; ++k) {
            float bv = -1e30f; int bi = 0;
            for (int t = 0; t < 64; ++t) {
                bool skip = false;
                for (int j = 0; j < k; ++j) if (chosen[j] == t) skip = true;
                if (!skip && row[t] > bv) { bv = row[t]; bi = t; }
            }
            chosen[k] = bi; idx_sh[tid * 3 + k] = bi;
        }
    }
    __syncthreads();
    if (tid < 8) {                                   // score_nor: normal rows 0..7
        float a = 0.f;
        for (int k = 0; k < 3; ++k) a += s[tid * 64 + idx_sh[tid * 3 + k]];
        out[8 + tid] = a * (1.f / 3.f);
    } else if (tid < 16) {                           // score_abn: abnormal rows 8..15
        int r = tid - 8;
        float a = 0.f;
        for (int k = 0; k < 3; ++k) a += s[(8 + r) * 64 + idx_sh[(8 + r) * 3 + k]];
        out[r] = a * (1.f / 3.f);
    }
    if (tid < 48) idxbuf[tid] = idx_sh[tid];
}

// Gather abn_fm / nor_fm (crop-major (ncrops*8, K=3, 1024))
__global__ __launch_bounds__(256)
void k_gather(const float* __restrict__ xl, const int* __restrict__ idxbuf,
              float* __restrict__ out)
{
    int blk = blockIdx.x;               // 480 = 2 * 10 * 8 * 3
    int isNor = blk >= 240;
    int r = blk % 240;
    int crop = r / 24;
    int b = (r / 3) % 8;
    int k = r % 3;
    int t, row; long long outBase;
    if (!isNor) {
        t = idxbuf[(8 + b) * 3 + k];
        row = (8 + b) * 10 + crop;
        outBase = 16 + (long long)((crop * 8 + b) * 3 + k) * 1024;
    } else {
        t = idxbuf[b * 3 + k];
        row = b * 10 + crop;
        outBase = 16 + 245760 + (long long)((crop * 8 + b) * 3 + k) * 1024;
    }
    const float* src = xl + ((long long)row * 64 + t) * 1024;
    for (int f = threadIdx.x; f < 1024; f += 256) out[outBase + f] = src[f];
}

// ==================================================================
extern "C" void kernel_launch(void* const* d_in, const int* in_sizes, int n_in,
                              void* d_out, int out_size, void* d_ws, size_t ws_size,
                              hipStream_t stream)
{
    const long long VIDEO_N = 16LL * 10 * 64 * 2049;
    const float* video; int pofs;
    if ((long long)in_sizes[0] == VIDEO_N) { video = (const float*)d_in[0]; pofs = 1; }
    else                                   { video = (const float*)d_in[n_in - 1]; pofs = 0; }
    auto P = [&](int i) { return (const float*)d_in[pofs + i]; };

    const int NB = 160, T = 64;
    const long long S   = 160LL * 1024 * 64;   // 10.5M floats per activation buffer
    const long long HSZ = 32LL * 4096 * 64;    // FF hidden chunk buffer
    float* buf0 = (float*)d_ws;
    float* buf1 = buf0 + S;
    float* buf2 = buf1 + S;
    float* bufH = buf2 + S;
    float* scrS = bufH + HSZ;                  // score_raw[10240]
    float* scrM = scrS + 10240;                // mag_raw[10240]
    int*   scrI = (int*)(scrM + 10240);        // top-k indices[48]

    float* x = buf0; float* t1 = buf1; float* t2 = buf2;

    // M % 64 == 0, K % 32 == 0 for every call below.
    auto gemm = [&](const float* A, int aRS, int aCS, int aTS, int taps, int pad,
                    const float* B, int bKS, int bNS, int bBS,
                    const float* bias, const float* add, int addBS,
                    float* Cp, int cBS, int M, int K, int batch,
                    float scale, int gelu) {
        dim3 grd(M / 64, batch, 1);
        k_gemm_wmma<<<grd, 128, 0, stream>>>(A, aRS, aCS, aTS, taps, pad,
                                             B, bKS, bNS, bBS, bias, add, addBS,
                                             Cp, cBS, M, K, scale, gelu);
    };

    // ---- stem: to_tokens (2048->64, k=3) reading video[b,t,c]; then += 0.1*to_mag
    gemm(P(139), 2048 * 3, 3, 1, 3, 1, video, 1, 2049, 64 * 2049,
         P(138), nullptr, 0, x, 64 * 64, 64, 2048, NB, 1.f, 0);
    k_magadd<<<(NB * 64 * 64 + 255) / 256, 256, 0, stream>>>(video, P(137), P(136), x, NB * 64 * 64);

    // ---- stage 0: dim 64, glance attention, 3 layers
    for (int l = 0; l < 3; ++l) {
        int pb = 2 + 13 * l;
        // scc (k=3) + residual
        gemm(P(pb + 12), 64 * 3, 3, 1, 3, 1, x, 64, 1, 64 * 64,
             P(pb + 11), x, 64 * 64, t1, 64 * 64, 64, 64, NB, 1.f, 0);
        std::swap(x, t1);
        // glance
        k_layernorm<<<NB * T, 256, 0, stream>>>(x, P(pb + 1), P(pb + 0), t1, 64, T, 0);
        gemm(P(pb + 4), 64, 1, 1, 1, 0, t1, 64, 1, 64 * 64,
             nullptr, nullptr, 0, t2, 192 * 64, 192, 64, NB, 1.f, 0);
        k_glance_attn<<<NB, 64, 0, stream>>>(t2, t1);
        gemm(P(pb + 3), 64, 1, 1, 1, 0, t1, 64, 1, 64 * 64,
             P(pb + 2), x, 64 * 64, t2, 64 * 64, 64, 64, NB, 1.f, 0);
        std::swap(x, t2);
        // ff
        k_layernorm<<<NB * T, 256, 0, stream>>>(x, P(pb + 8), P(pb + 7), t1, 64, T, 0);
        gemm(P(pb + 9), 64, 1, 1, 1, 0, t1, 64, 1, 64 * 64,
             P(pb + 5), nullptr, 0, bufH, 256 * 64, 256, 64, NB, 1.f, 1);
        gemm(P(pb + 10), 256, 1, 1, 1, 0, bufH, 64, 1, 256 * 64,
             P(pb + 6), x, 64 * 64, t2, 64 * 64, 64, 256, NB, 1.f, 0);
        std::swap(x, t2);
    }
    // trans 0: 64 -> 128
    k_layernorm<<<NB * T, 256, 0, stream>>>(x, P(43), P(42), t1, 64, T, 0);
    gemm(P(44), 64, 1, 1, 1, 0, t1, 64, 1, 64 * 64,
         P(41), nullptr, 0, t2, 128 * 64, 128, 64, NB, 1.f, 0);
    std::swap(x, t2);

    // ---- focus stages (1 and 2)
    auto fb_layer = [&](int pb, int dim, int heads, int ffh, int chunkB) {
        int cs = dim * T;
        long long tot = (long long)NB * cs;
        // scc (k=3) + residual
        gemm(P(pb + 16), dim * 3, 3, 1, 3, 1, x, T, 1, cs,
             P(pb + 15), x, cs, t1, cs, dim, dim, NB, 1.f, 0);
        std::swap(x, t1);
        // focus: BN -> v conv -> depthwise rel conv -> out conv + residual
        k_bn<<<(int)((tot + 255) / 256), 256, 0, stream>>>(x, P(pb + 1), P(pb + 0),
                                                           P(pb + 2), P(pb + 3), t1, dim, T, tot);
        gemm(P(pb + 8), dim, 1, 1, 1, 0, t1, T, 1, cs,
             nullptr, nullptr, 0, t2, cs, dim, dim, NB, 1.f, 0);
        k_dw<<<(int)((tot + 255) / 256), 256, 0, stream>>>(t2, P(pb + 7), P(pb + 6),
                                                           t1, dim, T, heads, tot);
        gemm(P(pb + 5), dim, 1, 1, 1, 0, t1, T, 1, cs,
             P(pb + 4), x, cs, t2, cs, dim, dim, NB, 1.f, 0);
        std::swap(x, t2);
        // ff (batch-chunked so the hidden tensor fits bufH)
        k_layernorm<<<NB * T, 256, 0, stream>>>(x, P(pb + 12), P(pb + 11), t1, dim, T, 0);
        int hs = ffh * T;
        for (int c0 = 0; c0 < NB; c0 += chunkB) {
            int nb = (NB - c0 < chunkB) ? (NB - c0) : chunkB;
            gemm(P(pb + 13), dim, 1, 1, 1, 0, t1 + (long long)c0 * cs, T, 1, cs,
                 P(pb + 9), nullptr, 0, bufH, hs, ffh, dim, nb, 1.f, 1);
            gemm(P(pb + 14), ffh, 1, 1, 1, 0, bufH, T, 1, hs,
                 P(pb + 10), x + (long long)c0 * cs, cs,
                 t2 + (long long)c0 * cs, cs, dim, ffh, nb, 1.f, 0);
        }
        std::swap(x, t2);
    };

    for (int l = 0; l < 3; ++l) fb_layer(45 + 17 * l, 128, 2, 512, 160);
    // trans 1: 128 -> 1024
    k_layernorm<<<NB * T, 256, 0, stream>>>(x, P(98), P(97), t1, 128, T, 0);
    gemm(P(99), 128, 1, 1, 1, 0, t1, T, 1, 128 * 64,
         P(96), nullptr, 0, t2, 1024 * 64, 1024, 128, NB, 1.f, 0);
    std::swap(x, t2);

    for (int l = 0; l < 2; ++l) fb_layer(100 + 17 * l, 1024, 16, 4096, 32);

    // ---- head: LN (var+eps form) + transpose to [bc,t,f], score/mag, msnsd
    k_layernorm<<<NB * T, 256, 0, stream>>>(x, P(135), P(134), t1, 1024, T, 1);
    k_score_mag<<<NB * T, 256, 0, stream>>>(t1, P(1), P(0), scrS, scrM, 1024);
    k_msnsd<<<1, 64, 0, stream>>>(scrS, scrM, (float*)d_out, scrI);
    k_gather<<<480, 256, 0, stream>>>(t1, scrI, (float*)d_out);

    (void)out_size; (void)ws_size;
}
